// PlasGraphModel_62912680952414
// MI455X (gfx1250) — matrix operand order repaired and checked by reading, not compiled, not run on hardware
//
#include <hip/hip_runtime.h>
#include <hip/hip_bf16.h>

typedef __attribute__((ext_vector_type(16))) _Float16 v16h;
typedef __attribute__((ext_vector_type(8)))  float    v8f;

#define N_NODES  100000
#define N_EDGES  3200000
#define F_IN     6
#define C_PRE    10
#define C        32
#define N_LAYERS 6
#define N_TILES  (N_NODES / 16)   // 6250, exact
#define TPW      5                // row-tiles per wave in GEMM kernels

static constexpr size_t AL(size_t x) { return (x + 255) & ~size_t(255); }

// ---------------- workspace layout (bytes) ----------------
static constexpr size_t OFF_DIS   = 0;                                       // N f32
static constexpr size_t OFF_NORM  = AL(OFF_DIS  + size_t(N_NODES) * 4);      // E f32
static constexpr size_t OFF_G     = AL(OFF_NORM + size_t(N_EDGES) * 4);      // N*32 f32
static constexpr size_t OFF_AGG   = AL(OFF_G    + size_t(N_NODES) * C * 4);  // N*32 f32
static constexpr size_t OFF_NI    = AL(OFF_AGG  + size_t(N_NODES) * C * 4);  // N*32 f16
static constexpr size_t OFF_H     = AL(OFF_NI   + size_t(N_NODES) * C * 2);  // N*32 f16
// pre-swizzled B fragments: [kstep][nb][lane][16 halves]
static constexpr size_t OFF_WGF   = AL(OFF_H    + size_t(N_NODES) * C * 2);  // 64*16 f16
static constexpr size_t OFF_WDF   = AL(OFF_WGF  + 64 * 16 * 2);              // 128*16 f16
static constexpr size_t OFF_WF1F  = AL(OFF_WDF  + 128 * 16 * 2);             // 128*16 f16

// ---------------- degree / normalization ----------------
__global__ void k_fill1(float* __restrict__ p, int n) {
    int i = blockIdx.x * blockDim.x + threadIdx.x;
    if (i < n) p[i] = 1.0f;
}

__global__ void k_deg(const long long* __restrict__ ei, float* __restrict__ deg) {
    int e = blockIdx.x * blockDim.x + threadIdx.x;
    if (e < N_EDGES) atomicAdd(&deg[(int)ei[N_EDGES + e]], 1.0f);
}

__global__ void k_rsqrt(float* __restrict__ d, int n) {
    int i = blockIdx.x * blockDim.x + threadIdx.x;
    if (i < n) d[i] = rsqrtf(d[i]);
}

__global__ void k_norm(const long long* __restrict__ ei, const float* __restrict__ dis,
                       float* __restrict__ norm) {
    int e = blockIdx.x * blockDim.x + threadIdx.x;
    if (e < N_EDGES) {
        int s = (int)ei[e];
        int d = (int)ei[N_EDGES + e];
        norm[e] = dis[s] * dis[d];
    }
}

// ------- weight prep: f32 [K,32] row-major -> f16 per-lane WMMA B fragments -------
// Fragment slot (kk*2+nb)*32+lane holds 16 halves: element j = W[kk*32 + (j>>3)*16 + hi*8 + (j&7)][nb*16 + l15]
__global__ void k_wprep(const float* __restrict__ Wg, const float* __restrict__ Wd,
                        const float* __restrict__ Wf1,
                        _Float16* __restrict__ WgF, _Float16* __restrict__ WdF,
                        _Float16* __restrict__ Wf1F) {
    int tid = blockIdx.x * blockDim.x + threadIdx.x;
    const float* src;
    _Float16* dst;
    int t;
    if (tid < 64)       { src = Wg;  dst = WgF;  t = tid; }        // K=32: 1 kstep
    else if (tid < 192) { src = Wd;  dst = WdF;  t = tid - 64; }   // K=64: 2 ksteps
    else if (tid < 320) { src = Wf1; dst = Wf1F; t = tid - 192; }
    else return;
    int kk   = t >> 6;          // 0 for Wg
    int nb   = (t >> 5) & 1;
    int lane = t & 31;
    int l15 = lane & 15, hi = lane >> 4;
#pragma unroll
    for (int j = 0; j < 16; ++j)
        dst[t * 16 + j] = (_Float16)src[(kk * 32 + (j >> 3) * 16 + hi * 8 + (j & 7)) * C + nb * 16 + l15];
}

// ---------------- preprocessing MLP (tiny, scalar) ----------------
__global__ void k_preproc(const float* __restrict__ x,
                          const float* __restrict__ Wpre, const float* __restrict__ bpre,
                          const float* __restrict__ Wfc1, const float* __restrict__ bfc1,
                          const float* __restrict__ Wfc2, const float* __restrict__ bfc2,
                          _Float16* __restrict__ ni16, _Float16* __restrict__ h16) {
    int n = blockIdx.x * blockDim.x + threadIdx.x;
    if (n >= N_NODES) return;
    float xi[F_IN];
#pragma unroll
    for (int k = 0; k < F_IN; ++k) xi[k] = x[n * F_IN + k];
    float h0[C_PRE];
#pragma unroll
    for (int c = 0; c < C_PRE; ++c) {
        float t = bpre[c];
#pragma unroll
        for (int k = 0; k < F_IN; ++k) t += xi[k] * Wpre[k * C_PRE + c];
        h0[c] = 1.0f / (1.0f + expf(-t));
    }
    for (int c = 0; c < C; ++c) {
        float t1 = bfc1[c], t2 = bfc2[c];
#pragma unroll
        for (int k = 0; k < C_PRE; ++k) {
            t1 += h0[k] * Wfc1[k * C + c];
            t2 += h0[k] * Wfc2[k * C + c];
        }
        ni16[n * C + c] = (_Float16)fmaxf(t1, 0.0f);
        h16[n * C + c]  = (_Float16)fmaxf(t2, 0.0f);
    }
}

// ---------------- GCN transform: g = h @ W; agg = g*selfnorm + bias ----------------
// B fragments pre-swizzled; each wave sweeps TPW row-tiles with B resident.
__global__ void k_gcn_gemm(const _Float16* __restrict__ A, const _Float16* __restrict__ BF,
                           const float* __restrict__ dis, const float* __restrict__ bias,
                           float* __restrict__ g, float* __restrict__ agg) {
    int gwave = blockIdx.x * (blockDim.x >> 5) + (threadIdx.x >> 5);
    int lane = threadIdx.x & 31;
    int l15 = lane & 15, hi = lane >> 4;

    union Frag { v16h v; uint4 q[2]; };
    Frag b[2];
#pragma unroll
    for (int nb = 0; nb < 2; ++nb) {
        const uint4* bp = reinterpret_cast<const uint4*>(BF + (nb * 32 + lane) * 16);
        b[nb].q[0] = bp[0];
        b[nb].q[1] = bp[1];
    }
    float bc[2] = { bias[l15], bias[16 + l15] };

    for (int t = 0; t < TPW; ++t) {
        int tile = gwave * TPW + t;
        if (tile >= N_TILES) break;                 // wave-uniform
        int rowA = tile * 16 + l15;

        Frag a;
        const _Float16* ap = A + rowA * C + hi * 8;
        a.q[0] = *reinterpret_cast<const uint4*>(ap);
        a.q[1] = *reinterpret_cast<const uint4*>(ap + 16);

        float selfn[8];
#pragma unroll
        for (int r = 0; r < 8; ++r) {
            float dv = dis[tile * 16 + r + hi * 8];
            selfn[r] = dv * dv;
        }

#pragma unroll
        for (int nb = 0; nb < 2; ++nb) {
            int col = nb * 16 + l15;
            v8f c = {};
            c = __builtin_amdgcn_wmma_f32_16x16x32_f16(false, a.v, false, b[nb].v,
                                                       (short)0, c, false, false);
#pragma unroll
            for (int r = 0; r < 8; ++r) {
                int row = tile * 16 + r + hi * 8;
                float gv = c[r];
                g[row * C + col]   = gv;
                agg[row * C + col] = gv * selfn[r] + bc[nb];
            }
        }
    }
}

// ---------------- edge scatter: agg[dst] += g[src] * norm (L2-resident atomics) ----
__global__ void k_scatter(const long long* __restrict__ ei, const float* __restrict__ norm,
                          const float* __restrict__ g, float* __restrict__ agg) {
    int tid = blockIdx.x * blockDim.x + threadIdx.x;
    if (tid >= N_EDGES * 4) return;
    int e = tid >> 2, q = tid & 3;
    int s = (int)ei[e];
    int d = (int)ei[N_EDGES + e];
    float w = norm[e];
    const float4* gs = reinterpret_cast<const float4*>(g + s * C + q * 8);
    float4 ga = gs[0];
    float4 gb = gs[1];
    float* ad = agg + d * C + q * 8;
    atomicAdd(ad + 0, ga.x * w);
    atomicAdd(ad + 1, ga.y * w);
    atomicAdd(ad + 2, ga.z * w);
    atomicAdd(ad + 3, ga.w * w);
    atomicAdd(ad + 4, gb.x * w);
    atomicAdd(ad + 5, gb.y * w);
    atomicAdd(ad + 6, gb.z * w);
    atomicAdd(ad + 7, gb.w * w);
}

// ------- dense layer, fused concat: out = relu([ni16, relu(agg)] @ W + b) -------
// A0 (K 0..31) from ni16 (f16); A1 (K 32..63) from agg (f32, relu+cvt in regs).
__global__ void k_dense_fused(const _Float16* __restrict__ NI, const float* __restrict__ AGG,
                              const _Float16* __restrict__ BF, const float* __restrict__ bias,
                              _Float16* __restrict__ out) {
    int gwave = blockIdx.x * (blockDim.x >> 5) + (threadIdx.x >> 5);
    int lane = threadIdx.x & 31;
    int l15 = lane & 15, hi = lane >> 4;

    union Frag { v16h v; uint4 q[2]; };
    Frag b[2][2];
#pragma unroll
    for (int kk = 0; kk < 2; ++kk)
#pragma unroll
        for (int nb = 0; nb < 2; ++nb) {
            const uint4* bp =
                reinterpret_cast<const uint4*>(BF + ((kk * 2 + nb) * 32 + lane) * 16);
            b[kk][nb].q[0] = bp[0];
            b[kk][nb].q[1] = bp[1];
        }
    float bc[2] = { bias[l15], bias[16 + l15] };

    for (int t = 0; t < TPW; ++t) {
        int tile = gwave * TPW + t;
        if (tile >= N_TILES) break;                 // wave-uniform
        int rowA = tile * 16 + l15;

        Frag a0;
        const _Float16* ap = NI + rowA * C + hi * 8;
        a0.q[0] = *reinterpret_cast<const uint4*>(ap);
        a0.q[1] = *reinterpret_cast<const uint4*>(ap + 16);

        v16h a1;
        const float* gp = AGG + rowA * C + hi * 8;
        float4 f0 = *reinterpret_cast<const float4*>(gp);
        float4 f1 = *reinterpret_cast<const float4*>(gp + 4);
        float4 f2 = *reinterpret_cast<const float4*>(gp + 16);
        float4 f3 = *reinterpret_cast<const float4*>(gp + 20);
        a1[0]  = (_Float16)fmaxf(f0.x, 0.0f);  a1[1]  = (_Float16)fmaxf(f0.y, 0.0f);
        a1[2]  = (_Float16)fmaxf(f0.z, 0.0f);  a1[3]  = (_Float16)fmaxf(f0.w, 0.0f);
        a1[4]  = (_Float16)fmaxf(f1.x, 0.0f);  a1[5]  = (_Float16)fmaxf(f1.y, 0.0f);
        a1[6]  = (_Float16)fmaxf(f1.z, 0.0f);  a1[7]  = (_Float16)fmaxf(f1.w, 0.0f);
        a1[8]  = (_Float16)fmaxf(f2.x, 0.0f);  a1[9]  = (_Float16)fmaxf(f2.y, 0.0f);
        a1[10] = (_Float16)fmaxf(f2.z, 0.0f);  a1[11] = (_Float16)fmaxf(f2.w, 0.0f);
        a1[12] = (_Float16)fmaxf(f3.x, 0.0f);  a1[13] = (_Float16)fmaxf(f3.y, 0.0f);
        a1[14] = (_Float16)fmaxf(f3.z, 0.0f);  a1[15] = (_Float16)fmaxf(f3.w, 0.0f);

#pragma unroll
        for (int nb = 0; nb < 2; ++nb) {
            int col = nb * 16 + l15;
            v8f c;
#pragma unroll
            for (int r = 0; r < 8; ++r) c[r] = bc[nb];
            c = __builtin_amdgcn_wmma_f32_16x16x32_f16(false, a0.v, false, b[0][nb].v,
                                                       (short)0, c, false, false);
            c = __builtin_amdgcn_wmma_f32_16x16x32_f16(false, a1, false, b[1][nb].v,
                                                       (short)0, c, false, false);
#pragma unroll
            for (int r = 0; r < 8; ++r) {
                int row = tile * 16 + r + hi * 8;
                out[row * C + col] = (_Float16)fmaxf(c[r], 0.0f);
            }
        }
    }
}

// ------- head dense layer: out = relu([ni16, h16] @ W + b), both sources f16 -------
__global__ void k_dense_head(const _Float16* __restrict__ NI, const _Float16* __restrict__ H,
                             const _Float16* __restrict__ BF, const float* __restrict__ bias,
                             _Float16* __restrict__ out) {
    int gwave = blockIdx.x * (blockDim.x >> 5) + (threadIdx.x >> 5);
    int lane = threadIdx.x & 31;
    int l15 = lane & 15, hi = lane >> 4;

    union Frag { v16h v; uint4 q[2]; };
    Frag b[2][2];
#pragma unroll
    for (int kk = 0; kk < 2; ++kk)
#pragma unroll
        for (int nb = 0; nb < 2; ++nb) {
            const uint4* bp =
                reinterpret_cast<const uint4*>(BF + ((kk * 2 + nb) * 32 + lane) * 16);
            b[kk][nb].q[0] = bp[0];
            b[kk][nb].q[1] = bp[1];
        }
    float bc[2] = { bias[l15], bias[16 + l15] };

    for (int t = 0; t < TPW; ++t) {
        int tile = gwave * TPW + t;
        if (tile >= N_TILES) break;                 // wave-uniform
        int rowA = tile * 16 + l15;

        Frag a0, a1;
        const _Float16* ap0 = NI + rowA * C + hi * 8;
        a0.q[0] = *reinterpret_cast<const uint4*>(ap0);
        a0.q[1] = *reinterpret_cast<const uint4*>(ap0 + 16);
        const _Float16* ap1 = H + rowA * C + hi * 8;
        a1.q[0] = *reinterpret_cast<const uint4*>(ap1);
        a1.q[1] = *reinterpret_cast<const uint4*>(ap1 + 16);

#pragma unroll
        for (int nb = 0; nb < 2; ++nb) {
            int col = nb * 16 + l15;
            v8f c;
#pragma unroll
            for (int r = 0; r < 8; ++r) c[r] = bc[nb];
            c = __builtin_amdgcn_wmma_f32_16x16x32_f16(false, a0.v, false, b[0][nb].v,
                                                       (short)0, c, false, false);
            c = __builtin_amdgcn_wmma_f32_16x16x32_f16(false, a1.v, false, b[1][nb].v,
                                                       (short)0, c, false, false);
#pragma unroll
            for (int r = 0; r < 8; ++r) {
                int row = tile * 16 + r + hi * 8;
                out[row * C + col] = (_Float16)fmaxf(c[r], 0.0f);
            }
        }
    }
}

// ---------------- final projection [N,32] @ [32,2] + b ----------------
__global__ void k_final(const _Float16* __restrict__ h, const float* __restrict__ Wf2,
                        const float* __restrict__ bf2, float* __restrict__ out) {
    int n = blockIdx.x * blockDim.x + threadIdx.x;
    if (n >= N_NODES) return;
    float a0 = bf2[0], a1 = bf2[1];
#pragma unroll
    for (int k = 0; k < C; ++k) {
        float hv = (float)h[n * C + k];
        a0 += hv * Wf2[k * 2 + 0];
        a1 += hv * Wf2[k * 2 + 1];
    }
    out[n * 2 + 0] = a0;
    out[n * 2 + 1] = a1;
}

extern "C" void kernel_launch(void* const* d_in, const int* in_sizes, int n_in,
                              void* d_out, int out_size, void* d_ws, size_t ws_size,
                              hipStream_t stream) {
    const float*     x    = (const float*)d_in[0];
    const long long* ei   = (const long long*)d_in[1];
    const float*     Wpre = (const float*)d_in[2];
    const float*     bpre = (const float*)d_in[3];
    const float*     Wfc1 = (const float*)d_in[4];
    const float*     bfc1 = (const float*)d_in[5];
    const float*     Wfc2 = (const float*)d_in[6];
    const float*     bfc2 = (const float*)d_in[7];
    const float*     Wgcn = (const float*)d_in[8];
    const float*     bgcn = (const float*)d_in[9];
    const float*     Wden = (const float*)d_in[10];
    const float*     bden = (const float*)d_in[11];
    const float*     Wf1  = (const float*)d_in[12];
    const float*     bf1  = (const float*)d_in[13];
    const float*     Wf2  = (const float*)d_in[14];
    const float*     bf2  = (const float*)d_in[15];
    float* out = (float*)d_out;

    char* ws = (char*)d_ws;
    float*     dis  = (float*)(ws + OFF_DIS);
    float*     norm = (float*)(ws + OFF_NORM);
    float*     g    = (float*)(ws + OFF_G);
    float*     agg  = (float*)(ws + OFF_AGG);
    _Float16*  ni16 = (_Float16*)(ws + OFF_NI);
    _Float16*  h16  = (_Float16*)(ws + OFF_H);
    _Float16*  WgF  = (_Float16*)(ws + OFF_WGF);
    _Float16*  WdF  = (_Float16*)(ws + OFF_WDF);
    _Float16*  Wf1F = (_Float16*)(ws + OFF_WF1F);

    const int TB = 256;
    const int gN  = (N_NODES + TB - 1) / TB;
    const int gE  = (N_EDGES + TB - 1) / TB;
    const int gE4 = (N_EDGES * 4 + TB - 1) / TB;
    const int WPB = 4;                                    // waves per block
    const int nWaves = (N_TILES + TPW - 1) / TPW;         // 1250
    const int gTile = (nWaves + WPB - 1) / WPB;           // 313

    // --- gcn_norm: deg -> rsqrt -> per-edge norm ---
    k_fill1<<<gN, TB, 0, stream>>>(dis, N_NODES);
    k_deg<<<gE, TB, 0, stream>>>(ei, dis);
    k_rsqrt<<<gN, TB, 0, stream>>>(dis, N_NODES);
    k_norm<<<gE, TB, 0, stream>>>(ei, dis, norm);

    // --- weights -> pre-swizzled f16 WMMA B fragments ---
    k_wprep<<<2, TB, 0, stream>>>(Wgcn, Wden, Wf1, WgF, WdF, Wf1F);

    // --- preprocessing MLP ---
    k_preproc<<<gN, TB, 0, stream>>>(x, Wpre, bpre, Wfc1, bfc1, Wfc2, bfc2, ni16, h16);

    // --- 6 tied GCN + dense layers (concat fused into dense GEMM) ---
    for (int l = 0; l < N_LAYERS; ++l) {
        k_gcn_gemm<<<gTile, WPB * 32, 0, stream>>>(h16, WgF, dis, bgcn, g, agg);
        k_scatter<<<gE4, TB, 0, stream>>>(ei, norm, g, agg);
        k_dense_fused<<<gTile, WPB * 32, 0, stream>>>(ni16, agg, WdF, bden, h16);
    }

    // --- head: fused concat(ni,h) @ Wf1 -> relu -> final 32x2 projection ---
    k_dense_head<<<gTile, WPB * 32, 0, stream>>>(ni16, h16, Wf1F, bf1, h16);
    k_final<<<gN, TB, 0, stream>>>(h16, Wf2, bf2, out);
}
